// LSTM_29420525977835
// MI455X (gfx1250) — compile-verified, matrix operand order checked by reading, not run on hardware
//
#include <hip/hip_runtime.h>
#include <hip/hip_bf16.h>

// ---------------------------------------------------------------------------
// LSTM on MI455X (gfx1250, wave32, WMMA, async global->LDS).
//   B=64, S=512, I=512, H=1024, 4H=4096.
// Phase 1: xg[BS,4H] = x[BS,I] @ WxT' + b     (bf16 WMMA, fp32 accum)
// Phase 2: 512x fused step kernels: g = xg_t + h @ WhT', gates, c/h update.
//          Wh in bf16 = 8MB -> resident in 192MB L2 across all 512 steps.
// K-loop is software-pipelined: async DMA of stage s+1 overlaps WMMA of s.
// ---------------------------------------------------------------------------

typedef __attribute__((ext_vector_type(8)))  float   v8f;
typedef __attribute__((ext_vector_type(16))) __bf16  v16bf;
typedef __attribute__((ext_vector_type(8)))  __bf16  v8bf;
typedef __attribute__((ext_vector_type(4)))  int     v4i;
typedef unsigned short u16;

#define BATCH 64
#define SEQ   512
#define IDIM  512
#define HDIM  1024
#define NG    4096            // 4*HDIM
#define MTOT  (BATCH*SEQ)     // 32768

// ---- async global->LDS (CDNA5), with synchronous fallback -----------------
#if defined(__AMDGCN__) && __has_builtin(__builtin_amdgcn_global_load_async_to_lds_b128)
#define HAS_ASYNC_LDS 1
#else
#define HAS_ASYNC_LDS 0
#endif

#if HAS_ASYNC_LDS
typedef __attribute__((address_space(1))) v4i gv4i;   // global int4
typedef __attribute__((address_space(3))) v4i lv4i;   // LDS int4
#define WAIT_ASYNC_3() asm volatile("s_wait_asynccnt 3" ::: "memory")
#define WAIT_ASYNC_0() asm volatile("s_wait_asynccnt 0" ::: "memory")
#else
#define WAIT_ASYNC_3()
#define WAIT_ASYNC_0()
#endif

// move one 16B chunk global -> LDS (async DMA if available)
__device__ __forceinline__ void cp16_g2l(const u16* g, __bf16* l) {
#if HAS_ASYNC_LDS
    __builtin_amdgcn_global_load_async_to_lds_b128((gv4i*)g, (lv4i*)l, 0, 0);
#else
    *(uint4*)l = *(const uint4*)g;
#endif
}

// round-to-nearest-even fp32 -> bf16 bits
__device__ __forceinline__ u16 f2bfu(float f) {
    unsigned u = __float_as_uint(f);
    unsigned r = u + 0x7fffu + ((u >> 16) & 1u);
    return (u16)(r >> 16);
}

// Gather one 16x32 bf16 fragment row from an 80B-padded LDS row.
// Per ISA 7.12.2 (16-bit A 16x32): lanes 0-15: K=0..7,16..23 ; lanes 16-31: K=8..15,24..31.
__device__ __forceinline__ v16bf frag_row(const __bf16* row /*40 elems*/, int hg) {
    v8bf lo = *(const v8bf*)(row + hg * 8);        // K = 8*hg .. 8*hg+7
    v8bf hi = *(const v8bf*)(row + 16 + hg * 8);   // K = 16+8*hg .. 23+8*hg
    return __builtin_shufflevector(lo, hi, 0,1,2,3,4,5,6,7,8,9,10,11,12,13,14,15);
}

// ------------------------- conversion / init kernels -----------------------

__global__ void cvt_x_kernel(const float* __restrict__ x, u16* __restrict__ xb) {
    int i4 = (blockIdx.x * 256 + threadIdx.x) * 4;   // MTOT*IDIM = 16777216 elems
    if (i4 < MTOT * IDIM) {
        xb[i4 + 0] = f2bfu(x[i4 + 0]);
        xb[i4 + 1] = f2bfu(x[i4 + 1]);
        xb[i4 + 2] = f2bfu(x[i4 + 2]);
        xb[i4 + 3] = f2bfu(x[i4 + 3]);
    }
}

// WxT[n][k] = X_{gate}[k][h] with n = gate*H + h   (N-major, K contiguous)
__global__ void cvt_wx_kernel(const float* __restrict__ X1, const float* __restrict__ X2,
                              const float* __restrict__ X3, const float* __restrict__ X4,
                              u16* __restrict__ wxT) {
    int idx = blockIdx.x * 256 + threadIdx.x;        // NG*IDIM = 2097152
    if (idx >= NG * IDIM) return;
    int n = idx >> 9, k = idx & (IDIM - 1);
    int gate = n >> 10, h = n & (HDIM - 1);
    const float* src = (gate == 0) ? X1 : (gate == 1) ? X2 : (gate == 2) ? X3 : X4;
    wxT[idx] = f2bfu(src[k * HDIM + h]);
}

// WhT[n][k] = H_{gate}[k][h]
__global__ void cvt_wh_kernel(const float* __restrict__ H1, const float* __restrict__ H2,
                              const float* __restrict__ H3, const float* __restrict__ H4,
                              u16* __restrict__ whT) {
    int idx = blockIdx.x * 256 + threadIdx.x;        // NG*HDIM = 4194304
    if (idx >= NG * HDIM) return;
    int n = idx >> 10, k = idx & (HDIM - 1);
    int gate = n >> 10, h = n & (HDIM - 1);
    const float* src = (gate == 0) ? H1 : (gate == 1) ? H2 : (gate == 2) ? H3 : H4;
    whT[idx] = f2bfu(src[k * HDIM + h]);
}

__global__ void cvt_bias_kernel(const float* __restrict__ b1, const float* __restrict__ b2,
                                const float* __restrict__ b3, const float* __restrict__ b4,
                                float* __restrict__ bcat) {
    int i = blockIdx.x * 256 + threadIdx.x;          // NG
    if (i >= NG) return;
    int gate = i >> 10, h = i & (HDIM - 1);
    const float* src = (gate == 0) ? b1 : (gate == 1) ? b2 : (gate == 2) ? b3 : b4;
    bcat[i] = src[h];
}

__global__ void init_state_kernel(float* __restrict__ cbuf, u16* __restrict__ hbf0) {
    int i = blockIdx.x * 256 + threadIdx.x;          // BATCH*HDIM = 65536
    if (i < BATCH * HDIM) { cbuf[i] = 0.0f; hbf0[i] = 0; }
}

// -------------------- phase 1: input projection GEMM -----------------------
// grid = (MTOT/64, NG/128), block = 256 (8 waves).
// Block tile 64x128; wave (mt = w&3, ng = w>>2) computes rows mt*16..+15,
// cols ng*64..+63 as 4 accumulator tiles. K step = 32 (bf16 WMMA).
// Double-buffered LDS stages; per wave 3 async ops outstanding per stage.

__global__ void __launch_bounds__(256)
gemm_xproj_kernel(const u16* __restrict__ xb, const u16* __restrict__ wxT,
                  const float* __restrict__ bcat, float* __restrict__ xg) {
    __shared__ alignas(16) __bf16 smA[2][64 * 40];    // 80B padded rows
    __shared__ alignas(16) __bf16 smB[2][128 * 40];

    const int tid  = threadIdx.x;
    const int wave = tid >> 5, lane = tid & 31;
    const int l16  = lane & 15, hg = lane >> 4;
    const int mt   = wave & 3,  ng = wave >> 2;
    const int m0   = blockIdx.x * 64, n0 = blockIdx.y * 128;
    const int rrA = tid >> 2, ccA = tid & 3;           // A-chunk assignment

    v8f acc[4] = {};

    auto issue_stage = [&](int step) {
        const int kk = step * 32, buf = step & 1;
        cp16_g2l(xb + (size_t)(m0 + rrA) * IDIM + kk + ccA * 8,
                 &smA[buf][rrA * 40 + ccA * 8]);
        #pragma unroll
        for (int i = 0; i < 2; ++i) {
            int it = tid + i * 256, rr = it >> 2, cc = it & 3;
            cp16_g2l(wxT + (size_t)(n0 + rr) * IDIM + kk + cc * 8,
                     &smB[buf][rr * 40 + cc * 8]);
        }
    };

    const int NSTEP = IDIM / 32;                       // 16
    issue_stage(0);
    for (int step = 0; step < NSTEP; ++step) {
        const int buf = step & 1;
        if (step + 1 < NSTEP) { issue_stage(step + 1); WAIT_ASYNC_3(); }
        else                  { WAIT_ASYNC_0(); }
        __syncthreads();                               // stage `buf` fully in LDS

        v16bf a = frag_row(&smA[buf][(mt * 16 + l16) * 40], hg);
        #pragma unroll
        for (int i = 0; i < 4; ++i) {
            v16bf b = frag_row(&smB[buf][(ng * 64 + i * 16 + l16) * 40], hg);
            acc[i] = __builtin_amdgcn_wmma_f32_16x16x32_bf16(
                         false, a, false, b, (short)0, acc[i], false, false);
        }
        __syncthreads();                               // done reading `buf`
    }

    #pragma unroll
    for (int i = 0; i < 4; ++i) {
        int n = n0 + ng * 64 + i * 16 + l16;
        float bias = bcat[n];
        #pragma unroll
        for (int r = 0; r < 8; ++r) {
            int m = m0 + mt * 16 + r + 8 * hg;    // C/D layout: lanes>=16 hold M+8
            xg[(size_t)m * NG + n] = acc[i][r] + bias;
        }
    }
}

// -------------------- phase 2: fused recurrent step ------------------------
// grid = 32 blocks; block j-range = [j0, j0+32). Block columns c in [0,128):
// gate = c>>5, n = gate*1024 + j0 + (c&31)  -> all 4 gates for this j-range
// live in one block, so gates fuse after an LDS exchange.

__global__ void __launch_bounds__(256)
lstm_step_kernel(const u16* __restrict__ hbf_in, const u16* __restrict__ whT,
                 const float* __restrict__ xg, float* __restrict__ cbuf,
                 float* __restrict__ out, u16* __restrict__ hbf_out, int t) {
    __shared__ alignas(16) __bf16 smA[2][64 * 40];
    __shared__ alignas(16) __bf16 smB[2][128 * 40];
    __shared__ float gmat[64 * 128];

    const int tid  = threadIdx.x;
    const int wave = tid >> 5, lane = tid & 31;
    const int l16  = lane & 15, hg = lane >> 4;
    const int mt   = wave & 3,  ng = wave >> 2;
    const int j0   = blockIdx.x * 32;
    const int rrA = tid >> 2, ccA = tid & 3;

    v8f acc[4] = {};

    auto issue_stage = [&](int step) {
        const int kk = step * 32, buf = step & 1;
        cp16_g2l(hbf_in + (size_t)rrA * HDIM + kk + ccA * 8,
                 &smA[buf][rrA * 40 + ccA * 8]);
        #pragma unroll
        for (int i = 0; i < 2; ++i) {
            int it = tid + i * 256, rr = it >> 2, cc = it & 3;
            int n = (rr >> 5) * HDIM + j0 + (rr & 31);   // gate-split weight rows
            cp16_g2l(whT + (size_t)n * HDIM + kk + cc * 8,
                     &smB[buf][rr * 40 + cc * 8]);
        }
    };

    const int NSTEP = HDIM / 32;                       // 32
    issue_stage(0);
    for (int step = 0; step < NSTEP; ++step) {
        const int buf = step & 1;
        if (step + 1 < NSTEP) { issue_stage(step + 1); WAIT_ASYNC_3(); }
        else                  { WAIT_ASYNC_0(); }
        __syncthreads();

        v16bf a = frag_row(&smA[buf][(mt * 16 + l16) * 40], hg);
        #pragma unroll
        for (int i = 0; i < 4; ++i) {
            v16bf b = frag_row(&smB[buf][(ng * 64 + i * 16 + l16) * 40], hg);
            acc[i] = __builtin_amdgcn_wmma_f32_16x16x32_bf16(
                         false, a, false, b, (short)0, acc[i], false, false);
        }
        __syncthreads();
    }

    // g = acc + xg[:, t, n]  -> LDS for the cross-gate exchange
    #pragma unroll
    for (int i = 0; i < 4; ++i) {
        int c = ng * 64 + i * 16 + l16;
        int n = (c >> 5) * HDIM + j0 + (c & 31);
        #pragma unroll
        for (int r = 0; r < 8; ++r) {
            int bI = mt * 16 + r + 8 * hg;     // batch index (M == BATCH)
            float g = acc[i][r] + xg[((size_t)bI * SEQ + t) * NG + n];
            gmat[bI * 128 + c] = g;
        }
    }
    __syncthreads();

    // gates: 64x32 (b, jj) pairs; 8 per thread
    #pragma unroll
    for (int p = 0; p < 8; ++p) {
        int idx = tid + p * 256;
        int bI = idx >> 5, jj = idx & 31;
        float f1 = gmat[bI * 128 + jj];          // forget
        float f2 = gmat[bI * 128 + 32 + jj];     // input
        float f3 = gmat[bI * 128 + 64 + jj];     // candidate
        float f4 = gmat[bI * 128 + 96 + jj];     // output
        float s1 = 1.0f / (1.0f + __expf(-f1));
        float s2 = 1.0f / (1.0f + __expf(-f2));
        float t1 = tanhf(f3);
        float s3 = 1.0f / (1.0f + __expf(-f4));
        int j  = j0 + jj;
        int ci = bI * HDIM + j;
        float cn = cbuf[ci] * s1 + s2 * t1;
        cbuf[ci] = cn;
        float h = tanhf(cn) * s3;
        out[((size_t)bI * SEQ + t) * HDIM + j] = h;
        hbf_out[ci] = f2bfu(h);
    }
}

// ------------------------------- launcher ----------------------------------

extern "C" void kernel_launch(void* const* d_in, const int* in_sizes, int n_in,
                              void* d_out, int out_size, void* d_ws, size_t ws_size,
                              hipStream_t stream) {
    const float* x  = (const float*)d_in[0];
    const float* X1 = (const float*)d_in[1];
    const float* H1 = (const float*)d_in[2];
    const float* b1 = (const float*)d_in[3];
    const float* X2 = (const float*)d_in[4];
    const float* H2 = (const float*)d_in[5];
    const float* b2 = (const float*)d_in[6];
    const float* X3 = (const float*)d_in[7];
    const float* H3 = (const float*)d_in[8];
    const float* b3 = (const float*)d_in[9];
    const float* X4 = (const float*)d_in[10];
    const float* H4 = (const float*)d_in[11];
    const float* b4 = (const float*)d_in[12];
    float* out = (float*)d_out;

    // workspace carve-up (all 16B aligned)
    char* ws = (char*)d_ws;
    float* xg   = (float*)(ws);                                   // 512 MB
    u16*   xb   = (u16*)(ws + (size_t)536870912);                 //  32 MB
    u16*   wxT  = (u16*)(ws + (size_t)570425344);                 //   4 MB
    u16*   whT  = (u16*)(ws + (size_t)574619648);                 //   8 MB
    float* bcat = (float*)(ws + (size_t)583008256);               //  16 KB
    float* cbuf = (float*)(ws + (size_t)583024640);               // 256 KB
    u16*   hbf0 = (u16*)(ws + (size_t)583286784);                 // 128 KB
    u16*   hbf1 = (u16*)(ws + (size_t)583417856);                 // 128 KB

    // pre-pass: bf16 conversion + weight transpose + state init
    cvt_x_kernel   <<<MTOT * IDIM / (256 * 4), 256, 0, stream>>>(x, xb);
    cvt_wx_kernel  <<<NG * IDIM / 256, 256, 0, stream>>>(X1, X2, X3, X4, wxT);
    cvt_wh_kernel  <<<NG * HDIM / 256, 256, 0, stream>>>(H1, H2, H3, H4, whT);
    cvt_bias_kernel<<<NG / 256, 256, 0, stream>>>(b1, b2, b3, b4, bcat);
    init_state_kernel<<<BATCH * HDIM / 256, 256, 0, stream>>>(cbuf, hbf0);

    // phase 1: xg = x @ WxT' + b  (WMMA bf16)
    gemm_xproj_kernel<<<dim3(MTOT / 64, NG / 128), 256, 0, stream>>>(xb, wxT, bcat, xg);

    // phase 2: sequential scan, h double-buffered in bf16
    for (int t = 0; t < SEQ; ++t) {
        const u16* hin = (t & 1) ? hbf1 : hbf0;
        u16*       hout = (t & 1) ? hbf0 : hbf1;
        lstm_step_kernel<<<NG / 128, 256, 0, stream>>>(hin, whT, xg, cbuf, out, hout, t);
    }
}
